// Attention_9036611191231
// MI455X (gfx1250) — matrix operand order repaired and checked by reading, not compiled, not run on hardware
//
#include <hip/hip_runtime.h>
#include <math.h>

typedef __attribute__((ext_vector_type(16))) __bf16 v16bf;
typedef __attribute__((ext_vector_type(8)))  __bf16 bf16x8;
typedef __attribute__((ext_vector_type(8)))  float  v8f;
typedef __attribute__((ext_vector_type(4)))  float  f32x4;

namespace {
constexpr int Bc = 4, Nc = 7, Lc = 512, Sc = 512, Hc = 8, Ec = 64, Dc = 64;
constexpr int WAVES         = 8;
constexpr int ROWS_PER_WAVE = 16;
constexpr int LTILE         = WAVES * ROWS_PER_WAVE;  // 128 rows of L per block
constexpr int LTILES        = Lc / LTILE;             // 4
constexpr int SBLK          = 32;                     // S columns per iteration
constexpr int KSTR          = 72;                     // K tile LDS row stride (bf16)
constexpr int VSTR          = 40;                     // V^T tile LDS row stride (bf16)
constexpr int PSTR          = 40;                     // P^T scratch LDS row stride (bf16)
}

__global__ __launch_bounds__(256, 2)
void fa_fwd_bf16_wmma_T(const float* __restrict__ Q,
                        const float* __restrict__ K,
                        const float* __restrict__ V,
                        float* __restrict__ O) {
    __shared__ __bf16 Kl[SBLK * KSTR];                   // K block, [s][e]
    __shared__ __bf16 Vl[Dc * VSTR];                     // V block transposed, [d][s]
    __shared__ __bf16 Pl[WAVES * ROWS_PER_WAVE * PSTR];  // per-wave P^T, [l][s]

    const int tid  = threadIdx.x;
    const int wave = tid >> 5;
    const int lane = tid & 31;
    const int nh   = lane & 15;          // N index (= this lane's L-row)
    const bool hi  = (lane >= 16);
    const int koff = hi ? 8 : 0;         // accumulator row offset (M = 8h + r) for this lane half
    const int boff = hi ? 16 : 0;        // K offset (k = 16h + s) of A and B in this lane half

    const int flat = blockIdx.x;
    const int lt   = flat % LTILES;
    const int bnh  = flat / LTILES;
    const int h    = bnh % Hc;
    const int bn   = bnh / Hc;

    const int rowBase = lt * LTILE + wave * ROWS_PER_WAVE;

    // ---- Q as B-matrices (scaled by 1/sqrt(E)): lane holds l-row rowBase+nh ----
    const float* qrow = Q + (((long)bn * Lc + rowBase + nh) * Hc + h) * Ec;
    v16bf qb[2];
#pragma unroll
    for (int ec = 0; ec < 2; ++ec) {
        const float* p0 = qrow + ec * 32 + boff;
        f32x4 x0 = *(const f32x4*)(p0);
        f32x4 x1 = *(const f32x4*)(p0 + 4);
        f32x4 x2 = *(const f32x4*)(p0 + 8);
        f32x4 x3 = *(const f32x4*)(p0 + 12);
#pragma unroll
        for (int i = 0; i < 4; ++i) {
            qb[ec][i]      = (__bf16)(x0[i] * 0.125f);
            qb[ec][4 + i]  = (__bf16)(x1[i] * 0.125f);
            qb[ec][8 + i]  = (__bf16)(x2[i] * 0.125f);
            qb[ec][12 + i] = (__bf16)(x3[i] * 0.125f);
        }
    }

    v8f oacc[4];                         // O^T accumulators: M=d (4 tiles), N=l
#pragma unroll
    for (int dt = 0; dt < 4; ++dt) oacc[dt] = (v8f)0.0f;
    float mrow = -__builtin_inff();
    float lrow = 0.0f;

    const float* Kbase = K + ((long)bn * Sc * Hc + h) * Ec;   // + s*Hc*Ec
    const float* Vbase = V + ((long)bn * Sc * Hc + h) * Dc;   // + s*Hc*Dc

    for (int s0 = 0; s0 < Sc; s0 += SBLK) {
        // ---- Stage K block as bf16 [s][e] (contiguous loads + b128 LDS store) ----
        {
            const int idx = tid * 8;             // 32*64 elements / 256 threads
            const int s = idx >> 6;
            const int e = idx & 63;
            const float* ksrc = Kbase + (long)(s0 + s) * Hc * Ec + e;
            f32x4 a = *(const f32x4*)ksrc;
            f32x4 b = *(const f32x4*)(ksrc + 4);
            bf16x8 w;
#pragma unroll
            for (int i = 0; i < 4; ++i) { w[i] = (__bf16)a[i]; w[4 + i] = (__bf16)b[i]; }
            *(bf16x8*)&Kl[s * KSTR + e] = w;
            if (s0 + SBLK < Sc)
                __builtin_prefetch(Kbase + (long)(s0 + SBLK + s) * Hc * Ec + e, 0, 1);
        }
        // ---- Stage V^T as bf16 [d][s] (coalesced global, b128 LDS store) ----
        {
            const int d  = tid & 63;
            const int sg = tid >> 6;             // 4 groups of 8 s-rows
            bf16x8 w;
#pragma unroll
            for (int i = 0; i < 8; ++i)
                w[i] = (__bf16)Vbase[(long)(s0 + sg * 8 + i) * Hc * Dc + d];
            *(bf16x8*)&Vl[d * VSTR + sg * 8] = w;
            if (s0 + SBLK < Sc)
                __builtin_prefetch(Vbase + (long)(s0 + SBLK + sg * 8) * Hc * Dc + d, 0, 1);
        }
        __syncthreads();

        // ---- S^T tiles: M = s (2 tiles of 16), N = l, K = e (2 WMMAs each) ----
        v8f tc[2];
#pragma unroll
        for (int cj = 0; cj < 2; ++cj) {
            v8f c = (v8f)0.0f;
#pragma unroll
            for (int ec = 0; ec < 2; ++ec) {
                const __bf16* ap = &Kl[(cj * 16 + nh) * KSTR + ec * 32 + boff];
                bf16x8 a0 = *(const bf16x8*)ap;
                bf16x8 a1 = *(const bf16x8*)(ap + 8);
                v16bf am;
#pragma unroll
                for (int i = 0; i < 8; ++i) { am[i] = a0[i]; am[8 + i] = a1[i]; }
                c = __builtin_amdgcn_wmma_f32_16x16x32_bf16(
                        false, am, false, qb[ec], (short)0, c, false, false);
            }
            tc[cj] = c;
        }

        // ---- Online softmax over s: in-lane tree + one cross-half shuffle ----
        float mx = tc[0][0];
#pragma unroll
        for (int cj = 0; cj < 2; ++cj)
#pragma unroll
            for (int r = 0; r < 8; ++r) mx = fmaxf(mx, tc[cj][r]);
        mx = fmaxf(mx, __shfl_xor(mx, 16, 32));
        const float mn    = fmaxf(mrow, mx);
        const float alpha = __expf(mrow - mn);
        mrow = mn;

        float rs = 0.0f;
        bf16x8 pw[2];
#pragma unroll
        for (int cj = 0; cj < 2; ++cj)
#pragma unroll
            for (int r = 0; r < 8; ++r) {
                const float p = __expf(tc[cj][r] - mn);
                rs += p;
                pw[cj][r] = (__bf16)p;
            }
        rs += __shfl_xor(rs, 16, 32);
        lrow = lrow * alpha + rs;

#pragma unroll
        for (int dt = 0; dt < 4; ++dt)
#pragma unroll
            for (int r = 0; r < 8; ++r) oacc[dt][r] *= alpha;

        // ---- Store P^T rows: lane's l-row, s = cj*16 + hi*8 + r (2x b128) ----
        __bf16* pb = &Pl[(wave * 16 + nh) * PSTR];
        *(bf16x8*)&pb[koff]      = pw[0];
        *(bf16x8*)&pb[16 + koff] = pw[1];

        // ---- Reload P^T as B-matrix (K = s = 32) ----
        const __bf16* pp = &Pl[(wave * 16 + nh) * PSTR + boff];
        bf16x8 pb0 = *(const bf16x8*)pp;
        bf16x8 pb1 = *(const bf16x8*)(pp + 8);
        v16bf pbm;
#pragma unroll
        for (int i = 0; i < 8; ++i) { pbm[i] = pb0[i]; pbm[8 + i] = pb1[i]; }

        // ---- O^T += V^T (A) x P^T (B) : 4 D-tiles ----
#pragma unroll
        for (int dt = 0; dt < 4; ++dt) {
            const __bf16* vp = &Vl[(dt * 16 + nh) * VSTR + boff];
            bf16x8 a0 = *(const bf16x8*)vp;
            bf16x8 a1 = *(const bf16x8*)(vp + 8);
            v16bf am;
#pragma unroll
            for (int i = 0; i < 8; ++i) { am[i] = a0[i]; am[8 + i] = a1[i]; }
            oacc[dt] = __builtin_amdgcn_wmma_f32_16x16x32_bf16(
                           false, am, false, pbm, (short)0, oacc[dt], false, false);
        }
        __syncthreads();
    }

    // ---- Epilogue: normalize, vectorized fp32 stores [B,N,L,H,D] ----
    const float inv = 1.0f / lrow;
    float* orow = O + (((long)bn * Lc + rowBase + nh) * Hc + h) * Dc;
#pragma unroll
    for (int dt = 0; dt < 4; ++dt) {
        f32x4 o0, o1;
#pragma unroll
        for (int i = 0; i < 4; ++i) { o0[i] = oacc[dt][i] * inv; o1[i] = oacc[dt][4 + i] * inv; }
        *(f32x4*)&orow[dt * 16 + koff * 1 + (hi ? 8 : 0) - koff] = o0;  // d = dt*16 + hi*8
        *(f32x4*)&orow[dt * 16 + (hi ? 8 : 0) + 4] = o1;
    }
}

extern "C" void kernel_launch(void* const* d_in, const int* in_sizes, int n_in,
                              void* d_out, int out_size, void* d_ws, size_t ws_size,
                              hipStream_t stream) {
    (void)in_sizes; (void)n_in; (void)d_ws; (void)ws_size; (void)out_size;
    const float* Q = (const float*)d_in[0];
    const float* K = (const float*)d_in[1];
    const float* V = (const float*)d_in[2];
    float* Out = (float*)d_out;

    const int grid = Bc * Nc * Hc * LTILES;  // 896 blocks
    fa_fwd_bf16_wmma_T<<<grid, WAVES * 32, 0, stream>>>(Q, K, V, Out);
}
